// HeteroGNN_84696755077152
// MI455X (gfx1250) — matrix operand order repaired and checked by reading, not compiled, not run on hardware
//
#include <hip/hip_runtime.h>

// ---------------- problem constants (match reference) ----------------
#define N_FLOWS 1000000
#define NF      76          // flow features
#define KP      96          // padded K for F=76 (3 x 32 chunks)
#define NH      50000       // hosts
#define NE      32          // host embedding dim
#define HID     64
#define NOUT    2
#define BN_EPS  1e-5f

typedef __bf16 bf16;
typedef __attribute__((ext_vector_type(16))) __bf16 v16bf;
typedef __attribute__((ext_vector_type(8)))  __bf16 v8bf;
typedef __attribute__((ext_vector_type(8)))  float  v8f;

// params layout (fp32, 576 entries)
#define PB1F  0
#define PSC1F 64
#define PSH1F 128
#define PB1H  192
#define PSC1H 256
#define PSH1H 320
#define PB2   384
#define PSC2  448
#define PSH2  512

// ---------------- CDNA5 async global->LDS copy (ASYNCcnt-tracked) --------
// LDS byte address = low 32 bits of the generic pointer (flat LDS aperture
// keeps the LDS offset in addr[31:0] per CDNA5 ISA 10.2).
__device__ __forceinline__ void async_copy_b128(bf16* lds_dst, const bf16* gsrc) {
  unsigned l = (unsigned)(size_t)lds_dst;
  unsigned long long g = (unsigned long long)(size_t)gsrc;
  asm volatile("global_load_async_to_lds_b128 %0, %1, off"
               :: "v"(l), "v"(g) : "memory");
}
__device__ __forceinline__ void wait_async0() {
  asm volatile("s_wait_asynccnt 0x0" ::: "memory");
}

// ---------------- WMMA helpers (CDNA5 16x16x32 bf16) ----------------
// A fragment (16x32 bf16): lane l (0..15) holds M=l, K = {0..7, 16..23};
// lane l (16..31) holds M=l-16, K = {8..15, 24..31}. 2 halfs per VGPR.
__device__ __forceinline__ v16bf load_a_frag(const bf16* a, int pitch, int tr,
                                             int kb, int lane) {
  int m    = lane & 15;
  int koff = kb + ((lane & 16) ? 8 : 0);
  const bf16* p = a + (size_t)(tr + m) * pitch + koff;
  union { v16bf v; v8bf h[2]; } u;
  u.h[0] = *(const v8bf*)(p);        // K koff .. koff+7
  u.h[1] = *(const v8bf*)(p + 16);   // K koff+16 .. koff+23
  return u.v;
}

// B fragment (32x16 bf16), weights stored transposed as Wt[n][k]:
// lanes 0..15: N=lane, K=kb+{0..7,16..23} ; lanes 16..31: N=lane-16, K=kb+{8..15,24..31}
__device__ __forceinline__ v16bf load_b_frag(const bf16* w, int pitch, int n0,
                                             int kb, int lane) {
  int n  = n0 + (lane & 15);
  int ks = kb + ((lane & 16) ? 8 : 0);
  const bf16* p = w + (size_t)n * pitch + ks;
  union { v16bf v; v8bf h[2]; } u;
  u.h[0] = *(const v8bf*)(p);
  u.h[1] = *(const v8bf*)(p + 16);
  return u.v;
}

__device__ __forceinline__ v8f wmma_bf16(v16bf a, v16bf b, v8f c) {
  return __builtin_amdgcn_wmma_f32_16x16x32_bf16(false, a, false, b,
                                                 (short)0, c, false, false);
}

// ---------------- kernel 0: pack weights (bf16, transposed) + BN fold ----
__global__ __launch_bounds__(256) void prep_kernel(
    const float* w1l_h2f, const float* w1r_h2f, const float* b1_flow,
    const float* w1l_f2h, const float* w1r_f2h, const float* b1_host,
    const float* w2l, const float* w2r, const float* b2,
    const float* bn1f, const float* bn1h, const float* bn2f,
    bf16* w1r_sum_t, bf16* wf2h0_t, bf16* wf2h1_t, bf16* wemb_t,
    bf16* w1l0_t, bf16* w1l1_t, bf16* w2r_t, bf16* w2l0_t, bf16* w2l1_t,
    float* params) {
  int tid = threadIdx.x;
  // F x HID weights, transposed + K padded to 96
  for (int i = tid; i < HID * KP; i += 256) {
    int n = i / KP, k = i % KP;
    float a = 0.f, c0 = 0.f, c1 = 0.f;
    if (k < NF) {
      a  = w1r_h2f[(size_t)k * HID + n] + w1r_h2f[(size_t)(NF * HID) + (size_t)k * HID + n];
      c0 = w1l_f2h[(size_t)k * HID + n];
      c1 = w1l_f2h[(size_t)(NF * HID) + (size_t)k * HID + n];
    }
    w1r_sum_t[(size_t)n * KP + k] = (bf16)a;
    wf2h0_t[(size_t)n * KP + k]   = (bf16)c0;
    wf2h1_t[(size_t)n * KP + k]   = (bf16)c1;
  }
  // E x HID weights, transposed (K=32 exact)
  for (int i = tid; i < HID * NE; i += 256) {
    int n = i / NE, k = i % NE;
    wemb_t[(size_t)n * NE + k] = (bf16)(w1r_f2h[(size_t)k * HID + n] +
                                        w1r_f2h[(size_t)(NE * HID) + (size_t)k * HID + n]);
    w1l0_t[(size_t)n * NE + k] = (bf16)w1l_h2f[(size_t)k * HID + n];
    w1l1_t[(size_t)n * NE + k] = (bf16)w1l_h2f[(size_t)(NE * HID) + (size_t)k * HID + n];
  }
  // HID x HID weights, transposed
  for (int i = tid; i < HID * HID; i += 256) {
    int n = i / HID, k = i % HID;
    w2r_t[(size_t)n * HID + k]  = (bf16)(w2r[(size_t)k * HID + n] +
                                         w2r[(size_t)(HID * HID) + (size_t)k * HID + n]);
    w2l0_t[(size_t)n * HID + k] = (bf16)w2l[(size_t)k * HID + n];
    w2l1_t[(size_t)n * HID + k] = (bf16)w2l[(size_t)(HID * HID) + (size_t)k * HID + n];
  }
  // BN fold + bias sums
  for (int c = tid; c < HID; c += 256) {
    params[PB1F + c] = b1_flow[c] + b1_flow[HID + c];
    params[PB1H + c] = b1_host[c] + b1_host[HID + c];
    params[PB2 + c]  = b2[c] + b2[HID + c];
    {
      float g = bn1f[c], b = bn1f[HID + c], rm = bn1f[2 * HID + c], rv = bn1f[3 * HID + c];
      float sc = g * rsqrtf(rv + BN_EPS);
      params[PSC1F + c] = sc; params[PSH1F + c] = b - rm * sc;
    }
    {
      float g = bn1h[c], b = bn1h[HID + c], rm = bn1h[2 * HID + c], rv = bn1h[3 * HID + c];
      float sc = g * rsqrtf(rv + BN_EPS);
      params[PSC1H + c] = sc; params[PSH1H + c] = b - rm * sc;
    }
    {
      float g = bn2f[c], b = bn2f[HID + c], rm = bn2f[2 * HID + c], rv = bn2f[3 * HID + c];
      float sc = g * rsqrtf(rv + BN_EPS);
      params[PSC2 + c] = sc; params[PSH2 + c] = b - rm * sc;
    }
  }
}

// ---------------- kernel 1: degree histogram ----------------
__global__ void hist_kernel(const int* src, const int* dst, int* cnt_src, int* cnt_dst) {
  int i = blockIdx.x * blockDim.x + threadIdx.x;
  if (i < N_FLOWS) {
    atomicAdd(&cnt_src[src[i]], 1);
    atomicAdd(&cnt_dst[dst[i]], 1);
  }
}

// ---------------- kernel 2: exclusive scan (block b = direction b) ------
__global__ __launch_bounds__(1024) void scan_kernel(const int* cnts, int* offs,
                                                    int* curs, float* invs) {
  int b = blockIdx.x;
  const int* cnt = cnts + (size_t)b * NH;
  int* off = offs + (size_t)b * (NH + 1);
  int* cur = curs + (size_t)b * NH;
  float* inv = invs + (size_t)b * NH;
  __shared__ int s[1024];
  __shared__ int carry;
  int tid = threadIdx.x;
  if (tid == 0) carry = 0;
  __syncthreads();
  for (int base = 0; base < NH; base += 1024) {
    int i = base + tid;
    int v = (i < NH) ? cnt[i] : 0;
    s[tid] = v;
    __syncthreads();
    for (int d = 1; d < 1024; d <<= 1) {
      int t = (tid >= d) ? s[tid - d] : 0;
      __syncthreads();
      s[tid] += t;
      __syncthreads();
    }
    int excl = s[tid] - v;
    if (i < NH) {
      off[i] = carry + excl;
      cur[i] = carry + excl;
      inv[i] = 1.0f / fmaxf((float)v, 1.0f);
    }
    __syncthreads();
    if (tid == 1023) carry += s[1023];
    __syncthreads();
  }
  if (tid == 0) off[NH] = carry;
}

// ---------------- kernel 3: scatter edge lists (CSR) ----------------
__global__ void scatter_kernel(const int* src, const int* dst, int* cur_src,
                               int* cur_dst, int* edge_src, int* edge_dst) {
  int i = blockIdx.x * blockDim.x + threadIdx.x;
  if (i < N_FLOWS) {
    int p = atomicAdd(&cur_src[src[i]], 1);
    edge_src[p] = i;
    int q = atomicAdd(&cur_dst[dst[i]], 1);
    edge_dst[q] = i;
  }
}

// ---------------- kernel 4: per-host mean aggregation (wave per task) ----
__global__ __launch_bounds__(256) void means_kernel(
    const float* flow_x, const int* edge_src, const int* edge_dst,
    const int* offs, const float* invs, bf16* mean_src, bf16* mean_dst) {
  int tid = threadIdx.x;
  int wid = tid >> 5, lane = tid & 31;
  int t = blockIdx.x * 8 + wid;
  if (t >= 2 * NH) return;
  int dir = (t >= NH) ? 1 : 0;
  int h = t - dir * NH;
  const int* edge = dir ? edge_dst : edge_src;
  const int* off  = offs + (size_t)dir * (NH + 1);
  float iv = invs[(size_t)dir * NH + h];
  int e0 = off[h], e1 = off[h + 1];
  float a0 = 0.f, a1 = 0.f, a2 = 0.f;
  int c2 = lane + 64;
  for (int e = e0; e < e1; ++e) {
    int fid = edge[e];
    const float* r = flow_x + (size_t)fid * NF;
    if (e + 1 < e1) __builtin_prefetch(flow_x + (size_t)edge[e + 1] * NF, 0, 0);
    a0 += r[lane];
    a1 += r[lane + 32];
    if (c2 < NF) a2 += r[c2];
  }
  bf16* m = (dir ? mean_dst : mean_src) + (size_t)h * KP;
  m[lane]      = (bf16)(a0 * iv);
  m[lane + 32] = (bf16)(a1 * iv);
  m[c2]        = (bf16)((c2 < NF) ? a2 * iv : 0.0f);  // lanes 12..31 zero the K-pad
}

// ---------------- kernel 5: host conv1  host_h = BNReLU(meanS@W0+meanD@W1+emb@We+b)
__global__ __launch_bounds__(256) void host_gemm1(
    const bf16* mean_src, const bf16* mean_dst, const float* emb,
    const bf16* wf2h0_t, const bf16* wf2h1_t, const bf16* wemb_t,
    const float* params, bf16* host_h) {
  __shared__ bf16 lS[128 * KP];
  __shared__ bf16 lD[128 * KP];
  __shared__ bf16 lE[128 * 40];
  int tid = threadIdx.x;
  int h0 = blockIdx.x * 128;
  // mean tiles: pure bf16 copies -> async DMA to LDS (ASYNCcnt)
  for (int i = tid; i < 128 * 12; i += 256) {
    int r = i / 12, c = i % 12;
    int h = h0 + r;
    bf16* lsd = lS + r * KP + c * 8;
    bf16* ldd = lD + r * KP + c * 8;
    if (h < NH) {
      async_copy_b128(lsd, mean_src + (size_t)h * KP + c * 8);
      async_copy_b128(ldd, mean_dst + (size_t)h * KP + c * 8);
    } else {
      v8bf z = {};
      *(v8bf*)lsd = z;
      *(v8bf*)ldd = z;
    }
  }
  for (int i = tid; i < 128 * NE; i += 256) {
    int r = i / NE, k = i % NE;
    int h = h0 + r;
    lE[r * 40 + k] = (h < NH) ? (bf16)emb[(size_t)h * NE + k] : (bf16)0.0f;
  }
  wait_async0();
  __syncthreads();
  int wid = tid >> 5, lane = tid & 31, tr = wid * 16;
  v8f acc[4] = {};
  for (int kc = 0; kc < 3; ++kc) {
    v16bf a = load_a_frag(lS, KP, tr, kc * 32, lane);
    for (int nt = 0; nt < 4; ++nt)
      acc[nt] = wmma_bf16(a, load_b_frag(wf2h0_t, KP, nt * 16, kc * 32, lane), acc[nt]);
    a = load_a_frag(lD, KP, tr, kc * 32, lane);
    for (int nt = 0; nt < 4; ++nt)
      acc[nt] = wmma_bf16(a, load_b_frag(wf2h1_t, KP, nt * 16, kc * 32, lane), acc[nt]);
  }
  {
    v16bf a = load_a_frag(lE, 40, tr, 0, lane);
    for (int nt = 0; nt < 4; ++nt)
      acc[nt] = wmma_bf16(a, load_b_frag(wemb_t, NE, nt * 16, 0, lane), acc[nt]);
  }
  for (int nt = 0; nt < 4; ++nt) {
    int col = nt * 16 + (lane & 15);
    float bias = params[PB1H + col], sc = params[PSC1H + col], sh = params[PSH1H + col];
    for (int j = 0; j < 8; ++j) {
      int r = tr + j + ((lane & 16) ? 8 : 0);
      int h = h0 + r;
      if (h < NH) {
        float v = fmaxf((acc[nt][j] + bias) * sc + sh, 0.0f);
        host_h[(size_t)h * HID + col] = (bf16)v;
      }
    }
  }
}

// ---------------- kernel 6: host projections (y=0: hp0/hp1, y=1: hsrc/hdst)
__global__ __launch_bounds__(256) void host_gemm2(
    const bf16* host_h, const float* emb,
    const bf16* w2l0_t, const bf16* w2l1_t,
    const bf16* w1l0_t, const bf16* w1l1_t,
    float* hp0, float* hp1, float* hsrc, float* hdst) {
  __shared__ bf16 lA[128 * 72];
  int tid = threadIdx.x;
  int h0 = blockIdx.x * 128;
  int mode = blockIdx.y;
  if (mode == 0) {
    for (int i = tid; i < 128 * 8; i += 256) {   // 8 x b128 chunks per 64-elem row
      int r = i >> 3, c = i & 7;
      int h = h0 + r;
      bf16* ldst = lA + r * 72 + c * 8;
      if (h < NH) {
        async_copy_b128(ldst, host_h + (size_t)h * HID + c * 8);
      } else {
        v8bf z = {};
        *(v8bf*)ldst = z;
      }
    }
  } else {
    for (int i = tid; i < 128 * NE; i += 256) {
      int r = i >> 5, k = i & 31;
      int h = h0 + r;
      lA[r * 40 + k] = (h < NH) ? (bf16)emb[(size_t)h * NE + k] : (bf16)0.0f;
    }
  }
  wait_async0();
  __syncthreads();
  int wid = tid >> 5, lane = tid & 31, tr = wid * 16;
  v8f a0[4] = {}, a1[4] = {};
  if (mode == 0) {
    for (int kc = 0; kc < 2; ++kc) {
      v16bf a = load_a_frag(lA, 72, tr, kc * 32, lane);
      for (int nt = 0; nt < 4; ++nt) {
        a0[nt] = wmma_bf16(a, load_b_frag(w2l0_t, HID, nt * 16, kc * 32, lane), a0[nt]);
        a1[nt] = wmma_bf16(a, load_b_frag(w2l1_t, HID, nt * 16, kc * 32, lane), a1[nt]);
      }
    }
  } else {
    v16bf a = load_a_frag(lA, 40, tr, 0, lane);
    for (int nt = 0; nt < 4; ++nt) {
      a0[nt] = wmma_bf16(a, load_b_frag(w1l0_t, NE, nt * 16, 0, lane), a0[nt]);
      a1[nt] = wmma_bf16(a, load_b_frag(w1l1_t, NE, nt * 16, 0, lane), a1[nt]);
    }
  }
  float* o0 = (mode == 0) ? hp0 : hsrc;
  float* o1 = (mode == 0) ? hp1 : hdst;
  for (int nt = 0; nt < 4; ++nt) {
    int col = nt * 16 + (lane & 15);
    for (int j = 0; j < 8; ++j) {
      int r = tr + j + ((lane & 16) ? 8 : 0);
      int h = h0 + r;
      if (h < NH) {
        o0[(size_t)h * HID + col] = a0[nt][j];
        o1[(size_t)h * HID + col] = a1[nt][j];
      }
    }
  }
}

// ---------------- kernel 7: flow pass 1 (conv1 flow branch) ----------------
__global__ __launch_bounds__(256) void flow_pass1(
    const float* flow_x, const int* src, const int* dst,
    const bf16* w1r_sum_t, const float* hsrc, const float* hdst,
    const float* params, bf16* flow_h) {
  __shared__ bf16 lA[128 * 104];
  __shared__ int sS[128];
  __shared__ int sD[128];
  int tid = threadIdx.x;
  size_t r0 = (size_t)blockIdx.x * 128;
  for (int i = tid; i < 128 * KP; i += 256) {
    int r = i / KP, k = i % KP;
    size_t fr = r0 + r;
    float v = (fr < N_FLOWS && k < NF) ? flow_x[fr * NF + k] : 0.0f;
    lA[r * 104 + k] = (bf16)v;
  }
  for (int i = tid; i < 128; i += 256) {
    size_t fr = r0 + i;
    sS[i] = (fr < N_FLOWS) ? src[fr] : 0;
    sD[i] = (fr < N_FLOWS) ? dst[fr] : 0;
  }
  __syncthreads();
  int wid = tid >> 5, lane = tid & 31, tr = wid * 16;
  v8f acc[4] = {};
  for (int kc = 0; kc < 3; ++kc) {
    v16bf a = load_a_frag(lA, 104, tr, kc * 32, lane);
    for (int nt = 0; nt < 4; ++nt)
      acc[nt] = wmma_bf16(a, load_b_frag(w1r_sum_t, KP, nt * 16, kc * 32, lane), acc[nt]);
  }
  for (int nt = 0; nt < 4; ++nt) {
    int col = nt * 16 + (lane & 15);
    float bias = params[PB1F + col], sc = params[PSC1F + col], sh = params[PSH1F + col];
    for (int j = 0; j < 8; ++j) {
      int rl = tr + j + ((lane & 16) ? 8 : 0);
      size_t fr = r0 + rl;
      if (fr < N_FLOWS) {
        int s = sS[rl], d = sD[rl];
        float v = acc[nt][j] + bias + hsrc[(size_t)s * HID + col] + hdst[(size_t)d * HID + col];
        v = fmaxf(v * sc + sh, 0.0f);
        flow_h[fr * HID + col] = (bf16)v;
      }
    }
  }
}

// ---------------- kernel 8: flow pass 2 (conv2 flow + BN + ReLU + linear) --
__global__ __launch_bounds__(256) void flow_pass2(
    const bf16* flow_h, const int* src, const int* dst, const bf16* w2r_t,
    const float* hp0, const float* hp1, const float* params,
    const float* lin_w, const float* lin_b, float* out) {
  __shared__ bf16 lA[128 * 72];
  __shared__ int sS[128];
  __shared__ int sD[128];
  __shared__ float h2[8][16 * 65];
  int tid = threadIdx.x;
  size_t r0 = (size_t)blockIdx.x * 128;
  // A tile: pure bf16 copy of flow_h rows -> async DMA to LDS (ASYNCcnt)
  for (int i = tid; i < 128 * 8; i += 256) {   // 8 x b128 chunks per 64-elem row
    int r = i >> 3, c = i & 7;
    size_t fr = r0 + r;
    bf16* ldst = lA + r * 72 + c * 8;
    if (fr < N_FLOWS) {
      async_copy_b128(ldst, flow_h + fr * HID + c * 8);
    } else {
      v8bf z = {};
      *(v8bf*)ldst = z;
    }
  }
  for (int i = tid; i < 128; i += 256) {
    size_t fr = r0 + i;
    sS[i] = (fr < N_FLOWS) ? src[fr] : 0;
    sD[i] = (fr < N_FLOWS) ? dst[fr] : 0;
  }
  wait_async0();
  __syncthreads();
  int wid = tid >> 5, lane = tid & 31, tr = wid * 16;
  v8f acc[4] = {};
  for (int kc = 0; kc < 2; ++kc) {
    v16bf a = load_a_frag(lA, 72, tr, kc * 32, lane);
    for (int nt = 0; nt < 4; ++nt)
      acc[nt] = wmma_bf16(a, load_b_frag(w2r_t, HID, nt * 16, kc * 32, lane), acc[nt]);
  }
  for (int nt = 0; nt < 4; ++nt) {
    int col = nt * 16 + (lane & 15);
    float bias = params[PB2 + col], sc = params[PSC2 + col], sh = params[PSH2 + col];
    for (int j = 0; j < 8; ++j) {
      int rl = j + ((lane & 16) ? 8 : 0);
      int row128 = tr + rl;
      int s = sS[row128], d = sD[row128];
      float v = acc[nt][j] + bias + hp0[(size_t)s * HID + col] + hp1[(size_t)d * HID + col];
      v = fmaxf(v * sc + sh, 0.0f);
      h2[wid][rl * 65 + col] = v;
    }
  }
  __syncthreads();
  // final linear: 64 -> 2 ; each lane owns (row = lane&15, oc = lane>>4)
  int row = lane & 15, oc = lane >> 4;
  float a = lin_b[oc];
  for (int k = 0; k < HID; ++k) a += h2[wid][row * 65 + k] * lin_w[k * NOUT + oc];
  size_t fr = r0 + tr + row;
  if (fr < N_FLOWS) out[fr * NOUT + oc] = a;
}

// ---------------- host: workspace layout + launch sequence ----------------
extern "C" void kernel_launch(void* const* d_in, const int* in_sizes, int n_in,
                              void* d_out, int out_size, void* d_ws, size_t ws_size,
                              hipStream_t stream) {
  const float* flow_x  = (const float*)d_in[0];
  const int*   src     = (const int*)d_in[1];
  const int*   dst     = (const int*)d_in[2];
  const float* emb     = (const float*)d_in[3];
  const float* w1l_h2f = (const float*)d_in[4];
  const float* w1r_h2f = (const float*)d_in[5];
  const float* b1_flow = (const float*)d_in[6];
  const float* w1l_f2h = (const float*)d_in[7];
  const float* w1r_f2h = (const float*)d_in[8];
  const float* b1_host = (const float*)d_in[9];
  const float* w2l     = (const float*)d_in[10];
  const float* w2r     = (const float*)d_in[11];
  const float* b2      = (const float*)d_in[12];
  const float* bn1f    = (const float*)d_in[13];
  const float* bn1h    = (const float*)d_in[14];
  const float* bn2f    = (const float*)d_in[15];
  const float* lin_w   = (const float*)d_in[16];
  const float* lin_b   = (const float*)d_in[17];
  float* out = (float*)d_out;

  char* ws = (char*)d_ws;
  size_t o = 0;
  auto take = [&](size_t bytes) -> char* {
    char* p = ws + o;
    o = (o + bytes + 255) & ~(size_t)255;
    return p;
  };

  bf16* w1r_sum_t = (bf16*)take((size_t)HID * KP * 2);
  bf16* wf2h0_t   = (bf16*)take((size_t)HID * KP * 2);
  bf16* wf2h1_t   = (bf16*)take((size_t)HID * KP * 2);
  bf16* wemb_t    = (bf16*)take((size_t)HID * NE * 2);
  bf16* w1l0_t    = (bf16*)take((size_t)HID * NE * 2);
  bf16* w1l1_t    = (bf16*)take((size_t)HID * NE * 2);
  bf16* w2r_t     = (bf16*)take((size_t)HID * HID * 2);
  bf16* w2l0_t    = (bf16*)take((size_t)HID * HID * 2);
  bf16* w2l1_t    = (bf16*)take((size_t)HID * HID * 2);
  float* params   = (float*)take(576 * 4);
  int* cnt        = (int*)take((size_t)2 * NH * 4);        // [src | dst]
  int* offs       = (int*)take((size_t)2 * (NH + 1) * 4);  // [src | dst]
  int* curs       = (int*)take((size_t)2 * NH * 4);
  float* invs     = (float*)take((size_t)2 * NH * 4);
  int* edge_src   = (int*)take((size_t)N_FLOWS * 4);
  int* edge_dst   = (int*)take((size_t)N_FLOWS * 4);
  bf16* mean_src  = (bf16*)take((size_t)NH * KP * 2);
  bf16* mean_dst  = (bf16*)take((size_t)NH * KP * 2);
  bf16* host_h    = (bf16*)take((size_t)NH * HID * 2);
  float* hsrc     = (float*)take((size_t)NH * HID * 4);
  float* hdst     = (float*)take((size_t)NH * HID * 4);
  float* hp0      = (float*)take((size_t)NH * HID * 4);
  float* hp1      = (float*)take((size_t)NH * HID * 4);
  bf16* flow_h    = (bf16*)take((size_t)N_FLOWS * HID * 2);
  (void)ws_size; (void)in_sizes; (void)n_in; (void)out_size;

  const int flowBlocks = (N_FLOWS + 255) / 256;       // 3907
  const int tileBlocks = (N_FLOWS + 127) / 128;       // 7813
  const int hostBlocks = (NH + 127) / 128;            // 391
  const int meanBlocks = (2 * NH + 7) / 8;            // 12500

  prep_kernel<<<1, 256, 0, stream>>>(w1l_h2f, w1r_h2f, b1_flow, w1l_f2h, w1r_f2h,
                                     b1_host, w2l, w2r, b2, bn1f, bn1h, bn2f,
                                     w1r_sum_t, wf2h0_t, wf2h1_t, wemb_t, w1l0_t,
                                     w1l1_t, w2r_t, w2l0_t, w2l1_t, params);
  hipMemsetAsync(cnt, 0, (size_t)2 * NH * 4, stream);
  hist_kernel<<<flowBlocks, 256, 0, stream>>>(src, dst, cnt, cnt + NH);
  scan_kernel<<<2, 1024, 0, stream>>>(cnt, offs, curs, invs);
  scatter_kernel<<<flowBlocks, 256, 0, stream>>>(src, dst, curs, curs + NH,
                                                 edge_src, edge_dst);
  means_kernel<<<meanBlocks, 256, 0, stream>>>(flow_x, edge_src, edge_dst, offs,
                                               invs, mean_src, mean_dst);
  host_gemm1<<<hostBlocks, 256, 0, stream>>>(mean_src, mean_dst, emb, wf2h0_t,
                                             wf2h1_t, wemb_t, params, host_h);
  host_gemm2<<<dim3(hostBlocks, 2), 256, 0, stream>>>(host_h, emb, w2l0_t, w2l1_t,
                                                      w1l0_t, w1l1_t, hp0, hp1,
                                                      hsrc, hdst);
  flow_pass1<<<tileBlocks, 256, 0, stream>>>(flow_x, src, dst, w1r_sum_t, hsrc,
                                             hdst, params, flow_h);
  flow_pass2<<<tileBlocks, 256, 0, stream>>>(flow_h, src, dst, w2r_t, hp0, hp1,
                                             params, lin_w, lin_b, out);
}
